// TimeframeInteractionHierarchy_9431748182488
// MI455X (gfx1250) — matrix operand order repaired and checked by reading, not compiled, not run on hardware
//
#include <hip/hip_runtime.h>
#include <hip/hip_bf16.h>

typedef __bf16 v16bf __attribute__((ext_vector_type(16)));
typedef float  v8f   __attribute__((ext_vector_type(8)));

#define NTF 11
#define BSZ 512
#define HDIM 128
#define BSPAN 64

// ---------------------------------------------------------------------------
// Kernel 1: pack W1/W2/W3 into split-bf16 (hi,lo) WMMA A-fragments in ws.
// A-matrix 16-bit layout (16x32): lane = (row&15) | (K&8 ? 16 : 0),
// element e = (K&7) + 8*(K>>4)   (K = k within 32-wide j-block)
// Fragment = [32 lanes][16 bf16] = 1KB. Index:
//   f = ((((d*128 + o)*8 + i_blk)*4 + j_blk)*2 + part)
// ---------------------------------------------------------------------------
__global__ __launch_bounds__(256) void pack_w_kernel(
    const float* __restrict__ W1, const float* __restrict__ W2,
    const float* __restrict__ W3, __bf16* __restrict__ ap)
{
    int idx = blockIdx.x * 256 + threadIdx.x;          // 3 * 2^21 total
    int d   = idx >> 21;
    int rem = idx & ((1 << 21) - 1);                   // o*16384 + i*128 + j
    int o = rem >> 14;
    int i = (rem >> 7) & 127;
    int j = rem & 127;
    const float* W = (d == 0) ? W1 : ((d == 1) ? W2 : W3);
    float w  = W[rem];
    __bf16 hi = (__bf16)w;
    __bf16 lo = (__bf16)(w - (float)hi);
    int iblk = i >> 4;
    int jb   = j >> 5;
    int k    = j & 31;
    int lane = (i & 15) | ((k & 8) ? 16 : 0);
    int e    = (k & 7) | ((k >> 4) << 3);
    int f    = (((d * 128 + o) * 8 + iblk) * 4 + jb) * 2;
    ap[(f + 0) * 512 + lane * 16 + e] = hi;
    ap[(f + 1) * 512 + lane * 16 + e] = lo;
}

// ---------------------------------------------------------------------------
// Kernel 2: main bilinear GEMM via v_wmma_f32_16x16x32_bf16 (split-bf16 x3).
// grid: 33 (q,d) x 8 o-chunks(16) x 8 b-chunks(64). block = 256 (8 waves).
// y[dd][q][b][o] = sum_{i,j} h[q,b,i] * W_d[o,i,j] * s[b,j],
//   s = h[q+d] (if valid) + h[q-d] (if valid)
// ---------------------------------------------------------------------------
__global__ __launch_bounds__(256) void bilinear_gemm_kernel(
    const float* __restrict__ h, const __bf16* __restrict__ ap,
    float* __restrict__ y)
{
    // B-matrix 16-bit layout (32x16): lane = (col&15) | (K&8 ? 16 : 0), e = (K&7) + 8*(K>>4)
    __shared__ __align__(32) __bf16 sB[32][32][16];    // [frag][lane][elem] 32KB
    __shared__ float sX[HDIM][BSPAN + 4];              // x1 transposed, padded 34KB

    int bid    = blockIdx.x;
    int bchunk = bid & 7;  bid >>= 3;
    int ochunk = bid & 7;  bid >>= 3;
    int q  = bid % NTF;
    int dd = bid / NTF;            // 0..2 -> distance dd+1
    int dist = dd + 1;
    int tid  = threadIdx.x;
    int b0   = bchunk * BSPAN;

    const float* hq  = h + ((long)q * BSZ + b0) * HDIM;
    const float* hn1 = (q + dist < NTF) ? h + ((long)(q + dist) * BSZ + b0) * HDIM : nullptr;
    const float* hn2 = (q >= dist)      ? h + ((long)(q - dist) * BSZ + b0) * HDIM : nullptr;

    for (int t = tid; t < BSPAN * HDIM; t += 256) {
        int bl = t >> 7, j = t & 127;
        sX[j][bl] = hq[bl * HDIM + j];
        float s = 0.f;
        if (hn1) s += hn1[bl * HDIM + j];
        if (hn2) s += hn2[bl * HDIM + j];
        __bf16 hi = (__bf16)s;
        __bf16 lo = (__bf16)(s - (float)hi);
        int bblk = bl >> 4;
        int jb   = j >> 5;
        int k    = j & 31;
        int lane = (bl & 15) | ((k & 8) ? 16 : 0);
        int e    = (k & 7) | ((k >> 4) << 3);
        int f    = (bblk * 4 + jb) * 2;
        sB[f][lane][e]     = hi;
        sB[f + 1][lane][e] = lo;
    }
    __syncthreads();

    int wave = tid >> 5, lane = tid & 31;
    int blc  = (lane & 15);
    int ihalf = (lane & 16) ? 8 : 0;

    for (int oo = 0; oo < 2; ++oo) {
        int o = ochunk * 16 + wave * 2 + oo;
        float part[4] = {0.f, 0.f, 0.f, 0.f};
        for (int iblk = 0; iblk < 8; ++iblk) {
            v16bf Ah[4], Al[4];
            long fb = (((long)(dd * 128 + o) * 8 + iblk) * 4) * 2;
#pragma unroll
            for (int jb = 0; jb < 4; ++jb) {
                Ah[jb] = *reinterpret_cast<const v16bf*>(ap + (fb + jb * 2) * 512 + lane * 16);
                Al[jb] = *reinterpret_cast<const v16bf*>(ap + (fb + jb * 2 + 1) * 512 + lane * 16);
            }
#pragma unroll
            for (int bblk = 0; bblk < 4; ++bblk) {
                v8f C = {};
#pragma unroll
                for (int jb = 0; jb < 4; ++jb) {
                    v16bf Bh = *reinterpret_cast<const v16bf*>(&sB[(bblk * 4 + jb) * 2][lane][0]);
                    v16bf Bl = *reinterpret_cast<const v16bf*>(&sB[(bblk * 4 + jb) * 2 + 1][lane][0]);
                    C = __builtin_amdgcn_wmma_f32_16x16x32_bf16(false, Ah[jb], false, Bh, (short)0, C, false, false);
                    C = __builtin_amdgcn_wmma_f32_16x16x32_bf16(false, Ah[jb], false, Bl, (short)0, C, false, false);
                    C = __builtin_amdgcn_wmma_f32_16x16x32_bf16(false, Al[jb], false, Bh, (short)0, C, false, false);
                }
                int ib = iblk * 16 + ihalf;
                int bl = bblk * 16 + blc;
                float acc = part[bblk];
#pragma unroll
                for (int r = 0; r < 8; ++r) acc += sX[ib + r][bl] * C[r];
                part[bblk] = acc;
            }
        }
#pragma unroll
        for (int bblk = 0; bblk < 4; ++bblk) {
            float v = part[bblk] + __shfl_xor(part[bblk], 16, 32);
            if (lane < 16) {
                y[(((long)dd * NTF + q) * BSZ + b0 + bblk * 16 + lane) * HDIM + o] = v;
            }
        }
    }
}

// ---------------------------------------------------------------------------
// Kernel 3: folded long-range tail.
// acc_lr[q,b,o] = alpha_q*(Wl h[q] + b_lr)[o] + (Wr z_q)[o]
//   z_q = sum_{d=4..10} V_d*(h[q+d]+h[q-d] valid), alpha_q = sum V_d*cnt
// ---------------------------------------------------------------------------
__global__ __launch_bounds__(128) void longrange_kernel(
    const float* __restrict__ h, const float* __restrict__ Wlr,
    const float* __restrict__ blr, const float* __restrict__ V3p,
    const float* __restrict__ decayp, float* __restrict__ acclr)
{
    __shared__ float sH[64][HDIM];
    __shared__ float sZ[64][HDIM];
    int q  = blockIdx.x >> 3;
    int b0 = (blockIdx.x & 7) * 64;
    float V3f = *V3p, decay = *decayp;
    float Vd[7];
    float alpha = 0.f;
    for (int d = 4; d < NTF; ++d) {
        float v = V3f * __expf(-decay * (float)(d - 3));
        Vd[d - 4] = v;
        alpha += v * (float)((q + d < NTF) + (q >= d));
    }
    int tid = threadIdx.x;
    for (int t = tid; t < 64 * HDIM; t += 128) {
        int bl = t >> 7, i = t & 127;
        sH[bl][i] = h[((long)q * BSZ + b0 + bl) * HDIM + i];
        float z = 0.f;
        for (int d = 4; d < NTF; ++d) {
            if (q + d < NTF) z += Vd[d - 4] * h[((long)(q + d) * BSZ + b0 + bl) * HDIM + i];
            if (q >= d)      z += Vd[d - 4] * h[((long)(q - d) * BSZ + b0 + bl) * HDIM + i];
        }
        sZ[bl][i] = z;
    }
    __syncthreads();
    int o = tid;
    float acc[64];
#pragma unroll
    for (int b = 0; b < 64; ++b) acc[b] = 0.f;
    for (int i = 0; i < HDIM; ++i) {
        float wl = Wlr[o * 256 + i] * alpha;
        float wr = Wlr[o * 256 + 128 + i];
#pragma unroll 8
        for (int b = 0; b < 64; ++b) acc[b] += wl * sH[b][i] + wr * sZ[b][i];
    }
    float bb = alpha * blr[o];
    for (int b = 0; b < 64; ++b)
        acclr[((long)q * BSZ + b0 + b) * HDIM + o] = acc[b] + bb;
}

// ---------------------------------------------------------------------------
// Kernel 4: combine + LayerNorm. One block per (q,b) row, 128 threads.
// ---------------------------------------------------------------------------
__global__ __launch_bounds__(128) void finalize_kernel(
    const float* __restrict__ h, const float* __restrict__ y,
    const float* __restrict__ acclr,
    const float* __restrict__ V1p, const float* __restrict__ V2p,
    const float* __restrict__ V3p,
    const float* __restrict__ b1, const float* __restrict__ b2,
    const float* __restrict__ b3,
    const float* __restrict__ lnw, const float* __restrict__ lnb,
    float* __restrict__ out)
{
    __shared__ float red[4];
    long row = blockIdx.x;          // 0 .. NTF*BSZ-1
    int q = (int)(row / BSZ);
    int o = threadIdx.x;
    float V[3] = {*V1p, *V2p, *V3p};
    const float* bias[3] = {b1, b2, b3};
    float x = h[row * HDIM + o] + acclr[row * HDIM + o];
#pragma unroll
    for (int dd = 0; dd < 3; ++dd) {
        int dist = dd + 1;
        float cnt = (float)((q + dist < NTF) + (q >= dist));
        x += V[dd] * (y[((long)dd * NTF * BSZ + row) * HDIM + o] + cnt * bias[dd][o]);
    }
    float s = x;
    for (int off = 16; off; off >>= 1) s += __shfl_xor(s, off, 32);
    if ((o & 31) == 0) red[o >> 5] = s;
    __syncthreads();
    float mu = (red[0] + red[1] + red[2] + red[3]) * (1.0f / HDIM);
    float dx = x - mu;
    __syncthreads();
    float s2 = dx * dx;
    for (int off = 16; off; off >>= 1) s2 += __shfl_xor(s2, off, 32);
    if ((o & 31) == 0) red[o >> 5] = s2;
    __syncthreads();
    float var = (red[0] + red[1] + red[2] + red[3]) * (1.0f / HDIM);
    out[row * HDIM + o] = dx * rsqrtf(var + 1e-5f) * lnw[o] + lnb[o];
}

// ---------------------------------------------------------------------------
extern "C" void kernel_launch(void* const* d_in, const int* in_sizes, int n_in,
                              void* d_out, int out_size, void* d_ws, size_t ws_size,
                              hipStream_t stream)
{
    const float* h    = (const float*)d_in[0];
    const float* V1   = (const float*)d_in[1];
    const float* V2   = (const float*)d_in[2];
    const float* V3   = (const float*)d_in[3];
    const float* Vdec = (const float*)d_in[4];
    const float* W1   = (const float*)d_in[5];
    const float* b1   = (const float*)d_in[6];
    const float* W2   = (const float*)d_in[7];
    const float* b2   = (const float*)d_in[8];
    const float* W3   = (const float*)d_in[9];
    const float* b3   = (const float*)d_in[10];
    const float* Wlr  = (const float*)d_in[11];
    const float* blr  = (const float*)d_in[12];
    const float* lnw  = (const float*)d_in[13];
    const float* lnb  = (const float*)d_in[14];
    float* out = (float*)d_out;

    // workspace layout
    const size_t AP_BYTES = (size_t)3 * 128 * 8 * 4 * 2 * 512 * sizeof(__bf16); // 24 MB
    const size_t Y_BYTES  = (size_t)3 * NTF * BSZ * HDIM * sizeof(float);       // 8.65 MB
    __bf16* ap    = (__bf16*)d_ws;
    float*  y     = (float*)((char*)d_ws + AP_BYTES);
    float*  acclr = (float*)((char*)d_ws + AP_BYTES + Y_BYTES);

    pack_w_kernel<<<(3 * 2097152) / 256, 256, 0, stream>>>(W1, W2, W3, ap);
    bilinear_gemm_kernel<<<33 * 8 * 8, 256, 0, stream>>>(h, ap, y);
    longrange_kernel<<<NTF * 8, 128, 0, stream>>>(h, Wlr, blr, V3, Vdec, acclr);
    finalize_kernel<<<NTF * BSZ, 128, 0, stream>>>(h, y, acclr, V1, V2, V3,
                                                   b1, b2, b3, lnw, lnb, out);
}
